// GNN_6090263626011
// MI455X (gfx1250) — matrix operand order repaired and checked by reading, not compiled, not run on hardware
//
#include <hip/hip_runtime.h>
#include <hip/hip_bf16.h>

// CDNA5 / gfx1250 wave32 WMMA types (probe-verified signatures)
typedef __attribute__((ext_vector_type(16))) __bf16       v16bf;
typedef __attribute__((ext_vector_type(8)))  float        v8f;
typedef __attribute__((ext_vector_type(4)))  unsigned int v4u;

#define DD   128  // embed dim
#define DD2  256  // hidden dim
#define PAD  8    // LDS row pad (bf16 elems = 16B) -> bank-conflict-free B frags

// Load a 16-half B fragment from padded LDS via two 16B loads (16B-aligned): k p..p+7, p+16..p+23.
__device__ __forceinline__ v16bf lds_frag(const __bf16* p) {
  union { v4u u[2]; v16bf v; } r;
  r.u[0] = *(const v4u*)p;
  r.u[1] = *(const v4u*)(p + 16);
  return r.v;
}

__device__ __forceinline__ void cvt_pack8(v16bf& a, int o,
                                          const float4& p, const float4& q) {
  a[o+0]=(__bf16)p.x; a[o+1]=(__bf16)p.y; a[o+2]=(__bf16)p.z; a[o+3]=(__bf16)p.w;
  a[o+4]=(__bf16)q.x; a[o+5]=(__bf16)q.y; a[o+6]=(__bf16)q.z; a[o+7]=(__bf16)q.w;
}

__device__ __forceinline__ float4 fma4(float s, const float4& a, const float4& b) {
  float4 r; r.x = s*a.x+b.x; r.y = s*a.y+b.y; r.z = s*a.z+b.z; r.w = s*a.w+b.w;
  return r;
}

__device__ __forceinline__ float4 bnrelu4(const float4& sc, const float4& t,
                                          const float4& sh) {
  float4 r;
  r.x = fmaxf(sc.x*t.x+sh.x, 0.f); r.y = fmaxf(sc.y*t.y+sh.y, 0.f);
  r.z = fmaxf(sc.z*t.z+sh.z, 0.f); r.w = fmaxf(sc.w*t.w+sh.w, 0.f);
  return r;
}

// ---------------------------------------------------------------------------
__global__ void embed_kernel(const int* __restrict__ tok,
                             const float* __restrict__ emb,
                             float* __restrict__ h, int total) {
  int i = blockIdx.x * blockDim.x + threadIdx.x;
  if (i < total) h[i] = emb[(size_t)tok[i >> 7] * DD + (i & (DD - 1))];
}

__global__ void zero_f32(float* __restrict__ p, int n) {
  int i = blockIdx.x * blockDim.x + threadIdx.x;
  if (i < n) p[i] = 0.0f;
}

// ---------------------------------------------------------------------------
// Fused: ea = edge_attr @ W + b ; msg = relu(h[src]+ea) ; agg[dst] += msg
// Block = 128 thr (4 waves). Wave = 16-edge x 128-col strip (8 WMMA col tiles).
__global__ __launch_bounds__(128)
void edge_gemm_scatter(const float* __restrict__ edge_attr, // [E][128]
                       const float* __restrict__ Wl,        // [128][128]
                       const float* __restrict__ bl,        // [128]
                       const int* __restrict__ srcIdx,
                       const int* __restrict__ dstIdx,
                       const float* __restrict__ h,         // [N][128]
                       float* __restrict__ agg,             // [N][128]
                       int E_) {
  __shared__ __bf16 sWt[DD][DD + PAD]; // transposed: [n][k], padded rows
  int tid = threadIdx.x;
  for (int idx = tid; idx < DD * DD; idx += 128) {
    int k = idx >> 7, n = idx & (DD - 1);
    sWt[n][k] = (__bf16)Wl[idx];
  }
  __syncthreads();

  int wave = tid >> 5, lane = tid & 31;
  int hs = lane >> 4;      // lane half: K interleave select / C row group
  int lr = lane & 15;      // A row in tile / C column in tile
  int e0 = blockIdx.x * 64 + wave * 16;
  if (e0 + 16 > E_) return;

  // A fragments (4 K-chunks) built with b128 loads, held in VGPRs, reused 8x.
  v16bf afrag[4];
  const float4* arow4 = (const float4*)(edge_attr + (size_t)(e0 + lr) * DD);
#pragma unroll
  for (int c = 0; c < 4; ++c) {
    int b4 = c * 8 + hs * 2;                    // (c*32 + hs*8)/4
    cvt_pack8(afrag[c], 0, arow4[b4],     arow4[b4 + 1]); // K base..base+7
    cvt_pack8(afrag[c], 8, arow4[b4 + 4], arow4[b4 + 5]); // K base+16..+23
  }

  for (int n0 = 0; n0 < DD; n0 += 16) {
    v8f acc = {};
#pragma unroll
    for (int c = 0; c < 4; ++c) {
      v16bf bfrag = lds_frag(&sWt[n0 + lr][c * 32 + hs * 8]);
      acc = __builtin_amdgcn_wmma_f32_16x16x32_bf16(
          false, afrag[c], false, bfrag, (short)0, acc, false, false);
    }
    int col = n0 + lr;
    float bias = bl[col];
    int ebase = e0 + hs * 8;
#pragma unroll
    for (int i = 0; i < 8; ++i) {
      int e = ebase + i;
      int s = srcIdx[e], d = dstIdx[e];
      float v = acc[i] + bias + h[(size_t)s * DD + col];
      v = fmaxf(v, 0.0f);
      atomicAdd(&agg[(size_t)d * DD + col], v);
    }
  }
}

// ---------------------------------------------------------------------------
// t = ((1+eps)*h + agg) @ W1 + b1   -> [N][256], grid.y selects 128-col half
__global__ __launch_bounds__(128)
void node_mlp1(const float* __restrict__ h, const float* __restrict__ agg,
               const float* __restrict__ W1l, // [128][256]
               const float* __restrict__ b1l, // [256]
               const float* __restrict__ epsl,
               float* __restrict__ t, int Nn) {
  __shared__ __bf16 sWt[DD][DD + PAD]; // [n][k] for this column half
  int tid = threadIdx.x;
  int ch = blockIdx.y;
  for (int idx = tid; idx < DD * DD; idx += 128) {
    int k = idx >> 7, n = idx & (DD - 1);
    sWt[n][k] = (__bf16)W1l[k * DD2 + ch * DD + n];
  }
  __syncthreads();

  int wave = tid >> 5, lane = tid & 31;
  int hs = lane >> 4, lr = lane & 15;
  int tile = blockIdx.x * 4 + wave;
  if (tile * 16 + 16 > Nn) return;
  int r0 = tile * 16;
  float ep = 1.0f + *epsl;

  v16bf afrag[4];
  const float4* hrow4 = (const float4*)(h   + (size_t)(r0 + lr) * DD);
  const float4* grow4 = (const float4*)(agg + (size_t)(r0 + lr) * DD);
#pragma unroll
  for (int c = 0; c < 4; ++c) {
    int b4 = c * 8 + hs * 2;
    cvt_pack8(afrag[c], 0, fma4(ep, hrow4[b4],     grow4[b4]),
                           fma4(ep, hrow4[b4 + 1], grow4[b4 + 1]));
    cvt_pack8(afrag[c], 8, fma4(ep, hrow4[b4 + 4], grow4[b4 + 4]),
                           fma4(ep, hrow4[b4 + 5], grow4[b4 + 5]));
  }

  for (int n0 = 0; n0 < DD; n0 += 16) {
    v8f acc = {};
#pragma unroll
    for (int c = 0; c < 4; ++c) {
      v16bf bfrag = lds_frag(&sWt[n0 + lr][c * 32 + hs * 8]);
      acc = __builtin_amdgcn_wmma_f32_16x16x32_bf16(
          false, afrag[c], false, bfrag, (short)0, acc, false, false);
    }
    int col = ch * DD + n0 + lr;
    float bias = b1l[col];
    int rbase = r0 + hs * 8;
#pragma unroll
    for (int i = 0; i < 8; ++i)
      t[(size_t)(rbase + i) * DD2 + col] = acc[i] + bias;
  }
}

// ---------------------------------------------------------------------------
// h2 = relu(scale1*t + shift1) @ W2 + b2  -> [N][128], grid.y selects 64-col half
__global__ __launch_bounds__(128)
void node_mlp2(const float* __restrict__ t,
               const float* __restrict__ scale1, const float* __restrict__ shift1,
               const float* __restrict__ W2l, // [256][128]
               const float* __restrict__ b2l, // [128]
               float* __restrict__ h2, int Nn) {
  __shared__ __bf16 sWt[64][DD2 + PAD]; // [n][k] for this 64-col half
  int tid = threadIdx.x;
  int ch = blockIdx.y;
  for (int idx = tid; idx < 64 * DD2; idx += 128) {
    int k = idx >> 6, n = idx & 63;
    sWt[n][k] = (__bf16)W2l[k * DD + ch * 64 + n];
  }
  __syncthreads();

  int wave = tid >> 5, lane = tid & 31;
  int hs = lane >> 4, lr = lane & 15;
  int tile = blockIdx.x * 4 + wave;
  if (tile * 16 + 16 > Nn) return;
  int r0 = tile * 16;

  v16bf afrag[8]; // K = 256 -> 8 chunks, held in VGPRs
  const float4* trow4 = (const float4*)(t + (size_t)(r0 + lr) * DD2);
  const float4* sc4   = (const float4*)scale1;
  const float4* sh4   = (const float4*)shift1;
#pragma unroll
  for (int c = 0; c < 8; ++c) {
    int b4 = c * 8 + hs * 2;
    cvt_pack8(afrag[c], 0,
              bnrelu4(sc4[b4],     trow4[b4],     sh4[b4]),
              bnrelu4(sc4[b4 + 1], trow4[b4 + 1], sh4[b4 + 1]));
    cvt_pack8(afrag[c], 8,
              bnrelu4(sc4[b4 + 4], trow4[b4 + 4], sh4[b4 + 4]),
              bnrelu4(sc4[b4 + 5], trow4[b4 + 5], sh4[b4 + 5]));
  }

  for (int n0 = 0; n0 < 64; n0 += 16) {
    v8f acc = {};
#pragma unroll
    for (int c = 0; c < 8; ++c) {
      v16bf bfrag = lds_frag(&sWt[n0 + lr][c * 32 + hs * 8]);
      acc = __builtin_amdgcn_wmma_f32_16x16x32_bf16(
          false, afrag[c], false, bfrag, (short)0, acc, false, false);
    }
    int col = ch * 64 + n0 + lr;
    float bias = b2l[col];
    int rbase = r0 + hs * 8;
#pragma unroll
    for (int i = 0; i < 8; ++i)
      h2[(size_t)(rbase + i) * DD + col] = acc[i] + bias;
  }
}

// ---------------------------------------------------------------------------
// BatchNorm helpers (training-mode, biased stats over axis 0)
__global__ void bn_stats(const float* __restrict__ X, float* __restrict__ sums,
                         int Nn, int C, int rowsPer) {
  int c = threadIdx.x; // blockDim.x == C
  int r0 = blockIdx.x * rowsPer;
  int rend = r0 + rowsPer; if (rend > Nn) rend = Nn;
  float s = 0.0f, sq = 0.0f;
  for (int r = r0; r < rend; ++r) {
    float v = X[(size_t)r * C + c];
    s += v; sq += v * v;
  }
  atomicAdd(&sums[c], s);
  atomicAdd(&sums[C + c], sq);
}

__global__ void bn_make(const float* __restrict__ sums,
                        const float* __restrict__ g, const float* __restrict__ b,
                        float* __restrict__ scale, float* __restrict__ shift,
                        int Nn, int C) {
  int c = threadIdx.x;
  if (c >= C) return;
  float invN = 1.0f / (float)Nn;
  float mean = sums[c] * invN;
  float var = sums[C + c] * invN - mean * mean;
  float inv = rsqrtf(var + 1e-5f);
  float sc = g[c] * inv;
  scale[c] = sc;
  shift[c] = b[c] - mean * sc;
}

__global__ void bn_apply(const float* __restrict__ X,
                         const float* __restrict__ scale,
                         const float* __restrict__ shift,
                         float* __restrict__ Y, int total, int C, int doRelu) {
  int i = blockIdx.x * blockDim.x + threadIdx.x;
  if (i >= total) return;
  int c = i & (C - 1);
  float v = scale[c] * X[i] + shift[c];
  if (doRelu) v = fmaxf(v, 0.0f);
  Y[i] = v;
}

// ---------------------------------------------------------------------------
extern "C" void kernel_launch(void* const* d_in, const int* in_sizes, int n_in,
                              void* d_out, int out_size, void* d_ws, size_t ws_size,
                              hipStream_t stream) {
  const int*   x_tokens  = (const int*)d_in[0];
  const int*   edge_idx  = (const int*)d_in[1];
  const float* edge_attr = (const float*)d_in[2];
  const float* emb       = (const float*)d_in[3];
  const float* ee_W      = (const float*)d_in[4];
  const float* ee_b      = (const float*)d_in[5];
  const float* W1        = (const float*)d_in[6];
  const float* b1        = (const float*)d_in[7];
  const float* bn1_g     = (const float*)d_in[8];
  const float* bn1_b     = (const float*)d_in[9];
  const float* W2        = (const float*)d_in[10];
  const float* b2        = (const float*)d_in[11];
  const float* eps_p     = (const float*)d_in[12];
  const float* bn_g      = (const float*)d_in[13];
  const float* bn_b      = (const float*)d_in[14];
  float* out = (float*)d_out;

  const int Nn = in_sizes[0];
  const int Ee = in_sizes[1] / 2;
  const int L  = 3;

  // workspace layout (fp32)
  float* h      = (float*)d_ws;                 // N*128
  float* agg    = h + (size_t)Nn * DD;          // N*128
  float* t      = agg + (size_t)Nn * DD;        // N*256
  float* h2     = t + (size_t)Nn * DD2;         // N*128
  float* sums   = h2 + (size_t)Nn * DD;         // 512
  float* scale1 = sums + 512;                   // 256
  float* shift1 = scale1 + 256;                 // 256

  const int* srcIdx = edge_idx;
  const int* dstIdx = edge_idx + Ee;

  embed_kernel<<<(Nn * DD + 255) / 256, 256, 0, stream>>>(x_tokens, emb, h, Nn * DD);

  const int rowTiles   = (Nn + 15) / 16;
  const int mlpBlocks  = (rowTiles + 3) / 4;
  const int edgeBlocks = (Ee + 63) / 64;
  const int statsRows  = 256;
  const int statsBlk   = (Nn + statsRows - 1) / statsRows;

  for (int l = 0; l < L; ++l) {
    zero_f32<<<(Nn * DD + 255) / 256, 256, 0, stream>>>(agg, Nn * DD);
    edge_gemm_scatter<<<edgeBlocks, 128, 0, stream>>>(
        edge_attr, ee_W + (size_t)l * DD * DD, ee_b + l * DD,
        srcIdx, dstIdx, h, agg, Ee);

    node_mlp1<<<dim3(mlpBlocks, 2), 128, 0, stream>>>(
        h, agg, W1 + (size_t)l * DD * DD2, b1 + l * DD2, eps_p + l, t, Nn);

    zero_f32<<<2, 256, 0, stream>>>(sums, 512);
    bn_stats<<<statsBlk, DD2, 0, stream>>>(t, sums, Nn, DD2, statsRows);
    bn_make<<<1, DD2, 0, stream>>>(sums, bn1_g + l * DD2, bn1_b + l * DD2,
                                   scale1, shift1, Nn, DD2);

    node_mlp2<<<dim3(mlpBlocks, 2), 128, 0, stream>>>(
        t, scale1, shift1, W2 + (size_t)l * DD2 * DD, b2 + l * DD, h2, Nn);

    zero_f32<<<2, 256, 0, stream>>>(sums, 512);
    bn_stats<<<statsBlk, DD, 0, stream>>>(h2, sums, Nn, DD, statsRows);
    bn_make<<<1, DD, 0, stream>>>(sums, bn_g + l * DD, bn_b + l * DD,
                                  scale1, shift1, Nn, DD);

    float* dstH = (l < L - 1) ? h : out;
    bn_apply<<<(Nn * DD + 255) / 256, 256, 0, stream>>>(
        h2, scale1, shift1, dstH, Nn * DD, DD, (l < L - 1) ? 1 : 0);
  }
}
